// get_model_56272661512467
// MI455X (gfx1250) — matrix operand order repaired and checked by reading, not compiled, not run on hardware
//
#include <hip/hip_runtime.h>
#include <hip/hip_bf16.h>

// ---------------------------------------------------------------------------
// PointNet++ MSG part-segmentation pipeline for gfx1250 (MI455X).
// All 1x1-conv MLPs run through v_wmma_f32_16x16x32_f16 (f16 operands, f32
// accumulation). Each wave computes a 16x64 output strip (4 WMMA tiles) so the
// streamed A operand is reused 4x from registers; the 64-column weight strip
// is staged per-K-chunk into double-buffered LDS, cooperatively by the whole
// block, using gfx1250 async-to-LDS copies (ASYNCcnt) when available.
// ---------------------------------------------------------------------------

typedef __attribute__((ext_vector_type(16))) _Float16 v16h;
typedef __attribute__((ext_vector_type(8)))  _Float16 v8h;
typedef __attribute__((ext_vector_type(8)))  float    v8f;
typedef __attribute__((ext_vector_type(4)))  int      v4i;

static inline int ru16(int x) { return (x + 15) & ~15; }
static inline int ru32(int x) { return (x + 31) & ~31; }
static inline int ru64i(int x) { return (x + 63) & ~63; }

#if defined(__has_builtin)
#if __has_builtin(__builtin_amdgcn_global_load_async_to_lds_b128)
#define HAVE_ASYNC_LDS 1
#endif
#endif
#ifndef HAVE_ASYNC_LDS
#define HAVE_ASYNC_LDS 0
#endif

__device__ __forceinline__ void stage_b16(const _Float16* src, _Float16* dst)
{
#if HAVE_ASYNC_LDS
  __builtin_amdgcn_global_load_async_to_lds_b128(
      (__attribute__((address_space(1))) v4i*)src,
      (__attribute__((address_space(3))) v4i*)dst,
      0, 0);
#else
  *(v8h*)dst = *(const v8h*)src;
#endif
}

__device__ __forceinline__ void wait_async_lds()
{
#if HAVE_ASYNC_LDS
#if __has_builtin(__builtin_amdgcn_s_wait_asynccnt)
  __builtin_amdgcn_s_wait_asynccnt(0);
#else
  asm volatile("s_wait_asynccnt 0x0" ::: "memory");
#endif
#endif
}

// ---------------------------------------------------------------------------
// WMMA GEMM:  Y[Mp,Np](f32) = X[Mp,Kp](f16) * W^T + bias
// W stored row-major [N,K] (B(k,n) = W[n*Kp + k]); Kp%32==0, Np%64==0.
// Block = 256 threads = 8 waves. blockIdx.y selects a 64-column strip;
// each wave owns one 16-row M tile and computes 4 N tiles (16x64 strip).
// The B strip chunk (64 cols x 32 k = 4KB f16) is staged into double-buffered
// LDS by all 256 threads (16B each) with async-to-LDS copies.
// Fragment layouts used here (contiguous k, same for A and B):
//   A 16x32 f16 : lane&15 -> row m; elems e -> K = (lane>>4)*16 + e
//   B 32x16 f16 : lane&15 -> col n; elems e -> K = (lane>>4)*16 + e
//   C/D 16x16 f32: n = lane&15, m = r + (lane>>4)*8
// ---------------------------------------------------------------------------
__global__ __launch_bounds__(256) void wmma_gemm_bias_kernel(
    const _Float16* __restrict__ X, const _Float16* __restrict__ Wt,
    const float* __restrict__ bias, float* __restrict__ Y,
    int Mp, int Kp, int Np, int Nreal)
{
  __shared__ _Float16 bstage[2][64 * 32];

  const int n0   = blockIdx.y * 64;
  const int wv   = threadIdx.x >> 5;
  const int tm   = blockIdx.x * 8 + wv;
  const bool active = (tm * 16) < Mp;
  const int lane = threadIdx.x & 31;
  const int half = lane >> 4;
  const int mn   = lane & 15;

  const _Float16* xrow = X + (size_t)(tm * 16 + mn) * Kp;

  // cooperative staging assignment: thread t copies 16B of row sn, chunk sq
  const int sn = threadIdx.x >> 2;
  const int sq = threadIdx.x & 3;
  const _Float16* wsrc = Wt + (size_t)(n0 + sn) * Kp + sq * 8;
  _Float16* sdst[2] = { &bstage[0][sn * 32 + sq * 8], &bstage[1][sn * 32 + sq * 8] };

  v8f acc0 = {}, acc1 = {}, acc2 = {}, acc3 = {};
  const int nk = Kp >> 5;

  stage_b16(wsrc, sdst[0]);                       // prologue: chunk 0 -> buf 0

  union Frag { v16h v; v8h h[2]; };
  for (int kt = 0; kt < nk; ++kt) {
    wait_async_lds();
    __syncthreads();                              // chunk kt visible in buf kt&1
    if (kt + 1 < nk)                              // stage next chunk into other buf
      stage_b16(wsrc + (size_t)(kt + 1) * 32, sdst[(kt + 1) & 1]);
    const _Float16* bb = bstage[kt & 1];
    if (active) {
      const int k0 = kt * 32;
      __builtin_prefetch(xrow + k0 + 128, 0, 0);  // stream-ahead for A
      Frag a, b0, b1, b2, b3;
      a.h[0] = *(const v8h*)(xrow + k0 + half * 16);
      a.h[1] = *(const v8h*)(xrow + k0 + half * 16 + 8);
      b0.h[0] = *(const v8h*)(bb + (size_t)(mn)      * 32 + half * 16);
      b0.h[1] = *(const v8h*)(bb + (size_t)(mn)      * 32 + half * 16 + 8);
      b1.h[0] = *(const v8h*)(bb + (size_t)(16 + mn) * 32 + half * 16);
      b1.h[1] = *(const v8h*)(bb + (size_t)(16 + mn) * 32 + half * 16 + 8);
      b2.h[0] = *(const v8h*)(bb + (size_t)(32 + mn) * 32 + half * 16);
      b2.h[1] = *(const v8h*)(bb + (size_t)(32 + mn) * 32 + half * 16 + 8);
      b3.h[0] = *(const v8h*)(bb + (size_t)(48 + mn) * 32 + half * 16);
      b3.h[1] = *(const v8h*)(bb + (size_t)(48 + mn) * 32 + half * 16 + 8);
      acc0 = __builtin_amdgcn_wmma_f32_16x16x32_f16(false, a.v, false, b0.v, (short)0, acc0, false, false);
      acc1 = __builtin_amdgcn_wmma_f32_16x16x32_f16(false, a.v, false, b1.v, (short)0, acc1, false, false);
      acc2 = __builtin_amdgcn_wmma_f32_16x16x32_f16(false, a.v, false, b2.v, (short)0, acc2, false, false);
      acc3 = __builtin_amdgcn_wmma_f32_16x16x32_f16(false, a.v, false, b3.v, (short)0, acc3, false, false);
    }
  }
  if (!active) return;

  v8f accs[4] = {acc0, acc1, acc2, acc3};
#pragma unroll
  for (int j = 0; j < 4; ++j) {
    int gn = n0 + j * 16 + mn;
    float bv = (gn < Nreal) ? bias[gn] : 0.0f;
#pragma unroll
    for (int r = 0; r < 8; ++r) {
      int gm = tm * 16 + half * 8 + r;
      Y[(size_t)gm * Np + gn] = accs[j][r] + bv;
    }
  }
}

// pack f32 weights [N,K] into padded f16 [Np,Kp]
__global__ void pack_weights_kernel(const float* __restrict__ W, int Nr, int Kr,
                                    _Float16* __restrict__ Wp, int Np, int Kp)
{
  size_t i = (size_t)blockIdx.x * blockDim.x + threadIdx.x;
  size_t total = (size_t)Np * Kp;
  if (i >= total) return;
  int n = (int)(i / Kp), k = (int)(i % Kp);
  Wp[i] = (n < Nr && k < Kr) ? (_Float16)W[(size_t)n * Kr + k] : (_Float16)0.0f;
}

// ---- BatchNorm statistics: zero -> coalesced partial sums (f32 atomics) ->
//      finalize into per-channel scale/shift ----
__global__ void zero_stats_kernel(float* __restrict__ a, float* __restrict__ b, int n)
{
  int i = blockIdx.x * blockDim.x + threadIdx.x;
  if (i < n) { a[i] = 0.0f; b[i] = 0.0f; }
}

__global__ void bn_partial_kernel(const float* __restrict__ Y, int M, int Np, int Nreal,
                                  float* __restrict__ sums, float* __restrict__ sqs)
{
  for (int c = threadIdx.x; c < Nreal; c += blockDim.x) {
    float s = 0.0f, s2 = 0.0f;
    for (int m = blockIdx.x; m < M; m += gridDim.x) {
      float v = Y[(size_t)m * Np + c];
      s += v; s2 += v * v;
    }
    atomicAdd(&sums[c], s);
    atomicAdd(&sqs[c], s2);
  }
}

__global__ void bn_finalize_kernel(const float* __restrict__ sums, const float* __restrict__ sqs,
                                   int M, int Nreal,
                                   const float* __restrict__ g, const float* __restrict__ bt,
                                   float* __restrict__ scale, float* __restrict__ shift)
{
  int n = blockIdx.x * blockDim.x + threadIdx.x;
  if (n >= Nreal) return;
  float mean = sums[n] / (float)M;
  float var  = sqs[n] / (float)M - mean * mean;
  float sc   = g[n] * rsqrtf(var + 1e-5f);
  scale[n] = sc;
  shift[n] = bt[n] - mean * sc;
}

// apply BN+ReLU, emit padded f16 activation [Mp, KpOut]
__global__ void bn_relu_store_f16_kernel(const float* __restrict__ Y, int M, int Np, int Nreal,
                                         const float* __restrict__ scale,
                                         const float* __restrict__ shift,
                                         _Float16* __restrict__ out, int Mp, int KpOut)
{
  size_t i = (size_t)blockIdx.x * blockDim.x + threadIdx.x;
  size_t total = (size_t)Mp * KpOut;
  if (i >= total) return;
  int m = (int)(i / KpOut), c = (int)(i % KpOut);
  float v = 0.0f;
  if (m < M && c < Nreal) {
    v = Y[(size_t)m * Np + c] * scale[c] + shift[c];
    v = fmaxf(v, 0.0f);
  }
  out[i] = (_Float16)v;
}

// apply BN+ReLU, emit dense f32 feature [M, Nreal]
__global__ void bn_relu_store_f32_kernel(const float* __restrict__ Y, int M, int Np, int Nreal,
                                         const float* __restrict__ scale,
                                         const float* __restrict__ shift,
                                         float* __restrict__ out)
{
  size_t i = (size_t)blockIdx.x * blockDim.x + threadIdx.x;
  size_t total = (size_t)M * Nreal;
  if (i >= total) return;
  int m = (int)(i / Nreal), c = (int)(i % Nreal);
  float v = Y[(size_t)m * Np + c] * scale[c] + shift[c];
  out[i] = fmaxf(v, 0.0f);
}

// ---------------------------------------------------------------------------
// Point-cloud plumbing
// ---------------------------------------------------------------------------

// xyz [B,3,N] -> [B,N,3]
__global__ void transpose_xyz_kernel(const float* __restrict__ in, int N,
                                     float* __restrict__ out)
{
  int i = blockIdx.x * blockDim.x + threadIdx.x;
  int total = 2 * N * 3;
  if (i >= total) return;
  int b = i / (N * 3);
  int rem = i % (N * 3);
  int n = rem / 3, c = rem % 3;
  out[i] = in[(size_t)b * 3 * N + (size_t)c * N + n];
}

// farthest point sampling: one block per batch, iterative argmax
__global__ void fps_kernel(const float* __restrict__ xyz, int N, int S,
                           int* __restrict__ idx)
{
  int b = blockIdx.x;
  const float* p = xyz + (size_t)b * N * 3;
  extern __shared__ float smem[];
  float* dist = smem;                    // N floats
  float* rv   = smem + N;                // blockDim floats
  int*   ri   = (int*)(rv + blockDim.x); // blockDim ints
  __shared__ int cur;
  for (int i = threadIdx.x; i < N; i += blockDim.x) dist[i] = 1e10f;
  if (threadIdx.x == 0) cur = 0;
  __syncthreads();
  for (int s = 0; s < S; ++s) {
    if (threadIdx.x == 0) idx[b * S + s] = cur;
    __syncthreads();
    int c = cur;
    float cx = p[c * 3 + 0], cy = p[c * 3 + 1], cz = p[c * 3 + 2];
    float best = -1.0f; int bi = 0;
    for (int i = threadIdx.x; i < N; i += blockDim.x) {
      float dx = p[i * 3 + 0] - cx;
      float dy = p[i * 3 + 1] - cy;
      float dz = p[i * 3 + 2] - cz;
      float d = dx * dx + dy * dy + dz * dz;
      float dm = fminf(dist[i], d);
      dist[i] = dm;
      if (dm > best) { best = dm; bi = i; }
    }
    rv[threadIdx.x] = best; ri[threadIdx.x] = bi;
    __syncthreads();
    for (int o = (int)blockDim.x >> 1; o > 0; o >>= 1) {
      if ((int)threadIdx.x < o && rv[threadIdx.x + o] > rv[threadIdx.x]) {
        rv[threadIdx.x] = rv[threadIdx.x + o];
        ri[threadIdx.x] = ri[threadIdx.x + o];
      }
      __syncthreads();
    }
    if (threadIdx.x == 0) cur = ri[0];
    __syncthreads();
  }
}

__global__ void gather_xyz_kernel(const float* __restrict__ xyz, const int* __restrict__ idx,
                                  int N, int S, float* __restrict__ out)
{
  int i = blockIdx.x * blockDim.x + threadIdx.x;
  if (i >= 2 * S) return;
  int b = i / S, s = i % S;
  int g = idx[b * S + s];
  const float* src = xyz + ((size_t)b * N + g) * 3;
  float* dst = out + (size_t)i * 3;
  dst[0] = src[0]; dst[1] = src[1]; dst[2] = src[2];
}

// ball query: first K in-radius indices (by index order), padded with the first
__global__ void ball_query_kernel(const float* __restrict__ xyz, const float* __restrict__ nxyz,
                                  int N, int S, float r2, int K, int* __restrict__ gidx)
{
  int i = blockIdx.x * blockDim.x + threadIdx.x;
  if (i >= 2 * S) return;
  int b = i / S;
  const float* q = nxyz + (size_t)i * 3;
  float cx = q[0], cy = q[1], cz = q[2];
  const float* p = xyz + (size_t)b * N * 3;
  int* row = gidx + (size_t)i * K;
  int cnt = 0, first = 0; bool found = false;
  for (int j = 0; j < N && cnt < K; ++j) {
    float dx = p[j * 3 + 0] - cx;
    float dy = p[j * 3 + 1] - cy;
    float dz = p[j * 3 + 2] - cz;
    float d = dx * dx + dy * dy + dz * dz;
    if (d <= r2) {
      if (!found) { first = j; found = true; }
      row[cnt++] = j;
    }
  }
  for (; cnt < K; ++cnt) row[cnt] = first;
}

// gather grouped features + relative xyz into padded f16 GEMM input rows
__global__ void group_pack_kernel(const float* __restrict__ xyz, const float* __restrict__ nxyz,
                                  const float* __restrict__ feat, int featC,
                                  const int* __restrict__ gidx, int N, int S, int K,
                                  int Cin, int Kp, _Float16* __restrict__ X, int Mp, int M)
{
  int i = blockIdx.x * blockDim.x + threadIdx.x;
  if (i >= Mp) return;
  _Float16* row = X + (size_t)i * Kp;
  if (i >= M) {
    for (int c = 0; c < Kp; ++c) row[c] = (_Float16)0.0f;
    return;
  }
  int j  = i % K;
  int bs = i / K;
  int b  = bs / S;
  int g  = gidx[(size_t)bs * K + j];
  const float* f = feat + ((size_t)b * N + g) * featC;
  for (int c = 0; c < featC; ++c) row[c] = (_Float16)f[c];
  const float* pg = xyz  + ((size_t)b * N + g) * 3;
  const float* pc = nxyz + (size_t)bs * 3;
  row[featC + 0] = (_Float16)(pg[0] - pc[0]);
  row[featC + 1] = (_Float16)(pg[1] - pc[1]);
  row[featC + 2] = (_Float16)(pg[2] - pc[2]);
  for (int c = Cin; c < Kp; ++c) row[c] = (_Float16)0.0f;
}

// max over the nsample axis of a padded f16 activation, write f32 features
__global__ void maxpool_kernel(const _Float16* __restrict__ act, int BS, int nsample,
                               int Kp, int C, float* __restrict__ feat, int featC, int coff)
{
  int i = blockIdx.x * blockDim.x + threadIdx.x;
  if (i >= BS * C) return;
  int bs = i / C, c = i % C;
  const _Float16* p = act + (size_t)bs * nsample * Kp + c;
  float m = -1e30f;
  for (int j = 0; j < nsample; ++j) m = fmaxf(m, (float)p[(size_t)j * Kp]);
  feat[(size_t)bs * featC + coff + c] = m;
}

// 3-NN inverse-distance interpolation + concat(points1, interp) -> padded f16
__global__ void fp_interp_pack_kernel(const float* __restrict__ xyz1, const float* __restrict__ xyz2,
                                      const float* __restrict__ pts1, int C1,
                                      const float* __restrict__ pts2, int C2,
                                      int N1, int N2, _Float16* __restrict__ X,
                                      int Kp, int Mp, int M)
{
  int i = blockIdx.x * blockDim.x + threadIdx.x;
  if (i >= Mp) return;
  _Float16* row = X + (size_t)i * Kp;
  if (i >= M) {
    for (int c = 0; c < Kp; ++c) row[c] = (_Float16)0.0f;
    return;
  }
  int b = i / N1, n = i % N1;
  const float* q = xyz1 + ((size_t)b * N1 + n) * 3;
  const float* p2 = xyz2 + (size_t)b * N2 * 3;
  float d0 = 1e30f, d1 = 1e30f, d2 = 1e30f;
  int i0 = 0, i1 = 0, i2 = 0;
  for (int j = 0; j < N2; ++j) {
    float dx = p2[j * 3 + 0] - q[0];
    float dy = p2[j * 3 + 1] - q[1];
    float dz = p2[j * 3 + 2] - q[2];
    float d = dx * dx + dy * dy + dz * dz;
    if (d < d0)      { d2 = d1; i2 = i1; d1 = d0; i1 = i0; d0 = d; i0 = j; }
    else if (d < d1) { d2 = d1; i2 = i1; d1 = d;  i1 = j; }
    else if (d < d2) { d2 = d;  i2 = j; }
  }
  float w0 = 1.0f / (d0 + 1e-8f);
  float w1 = 1.0f / (d1 + 1e-8f);
  float w2 = 1.0f / (d2 + 1e-8f);
  float ws = w0 + w1 + w2;
  w0 /= ws; w1 /= ws; w2 /= ws;
  int c = 0;
  const float* f1p = pts1 + ((size_t)b * N1 + n) * C1;
  for (; c < C1; ++c) row[c] = (_Float16)f1p[c];
  const float* f0 = pts2 + ((size_t)b * N2 + i0) * C2;
  const float* f1 = pts2 + ((size_t)b * N2 + i1) * C2;
  const float* f2 = pts2 + ((size_t)b * N2 + i2) * C2;
  for (int k = 0; k < C2; ++k, ++c)
    row[c] = (_Float16)(w0 * f0[k] + w1 * f1[k] + w2 * f2[k]);
  for (; c < Kp; ++c) row[c] = (_Float16)0.0f;
}

// p0 = concat(cls broadcast, l0_xyz, l0_points) -> [B,N,7]
__global__ void build_p0_kernel(const float* __restrict__ cls, const float* __restrict__ xyz,
                                int N, float* __restrict__ p0)
{
  int i = blockIdx.x * blockDim.x + threadIdx.x;
  if (i >= 2 * N) return;
  int b = i / N;
  const float* x = xyz + (size_t)i * 3;
  float* r = p0 + (size_t)i * 7;
  r[0] = cls[b];
  r[1] = x[0]; r[2] = x[1]; r[3] = x[2];
  r[4] = x[0]; r[5] = x[1]; r[6] = x[2];
}

// head: logits = feat @ W2^T + b2 (2 classes), log_softmax
__global__ void head_logits_kernel(const _Float16* __restrict__ feat, int Kp,
                                   const float* __restrict__ W2, const float* __restrict__ b2,
                                   int M, float* __restrict__ out)
{
  int i = blockIdx.x * blockDim.x + threadIdx.x;
  if (i >= M) return;
  const _Float16* f = feat + (size_t)i * Kp;
  float l0 = b2[0], l1 = b2[1];
  for (int c = 0; c < 128; ++c) {
    float v = (float)f[c];
    l0 += v * W2[c];
    l1 += v * W2[128 + c];
  }
  float mx = fmaxf(l0, l1);
  float lse = mx + logf(expf(l0 - mx) + expf(l1 - mx));
  out[(size_t)i * 2 + 0] = l0 - lse;
  out[(size_t)i * 2 + 1] = l1 - lse;
}

// [B,S,C] -> [B,C,S]
__global__ void transpose_feat_kernel(const float* __restrict__ in, int S, int C,
                                      float* __restrict__ out)
{
  int i = blockIdx.x * blockDim.x + threadIdx.x;
  int total = 2 * S * C;
  if (i >= total) return;
  int b = i / (S * C);
  int rem = i % (S * C);
  int c = rem / S, s = rem % S;
  out[i] = in[((size_t)b * S + s) * C + c];
}

// ---------------------------------------------------------------------------
// Host-side orchestration
// ---------------------------------------------------------------------------
namespace {

struct LayerP { const float *w, *b, *g, *bt; int cin, cout; };

const int   SA_NPOINT[4]   = {1024, 512, 256, 128};
const float SA_RADII[4][5] = {{0.005f, 0.01f, 0.02f, 0.03f, 0.06f},
                              {0.02f, 0.03f, 0.06f, 0.12f, 0.24f},
                              {0.06f, 0.12f, 0.24f, 0.48f, 0.0f},
                              {0.06f, 0.12f, 0.24f, 0.48f, 0.0f}};
const int   SA_NS[4][5]    = {{32, 96, 192, 256, 384},
                              {24, 64, 96, 128, 196},
                              {28, 36, 48, 96, 0},
                              {14, 18, 24, 48, 0}};
const int   SA_NBR[4]      = {5, 5, 4, 4};
const int   SA_CIN[4]      = {6, 515, 1283, 1027};
const int   SA_DIMS[4][5][3] = {
  {{32,32,64},{32,32,64},{64,64,128},{64,96,128},{64,96,128}},
  {{128,128,256},{128,128,256},{128,128,256},{128,196,256},{128,196,256}},
  {{128,128,256},{128,196,256},{128,196,256},{128,196,256},{0,0,0}},
  {{128,128,256},{128,196,256},{128,196,256},{128,196,256},{0,0,0}}};
const int   FP_CIN[5]     = {2048, 1280, 1536, 640, 135};
const int   FP_DIMS[5][2] = {{512,256},{256,256},{256,128},{256,128},{128,128}};

struct Scratch {
  _Float16* Wp; float* Y; float* sc; float* sh; float* sums; float* sqs;
};

void run_layer(const LayerP& L, const _Float16* Xin, int M, int Mp,
               const Scratch& W, _Float16* Xout, hipStream_t st)
{
  int Kp = ru32(L.cin), Np = ru64i(L.cout), KpO = ru32(L.cout);
  size_t wn = (size_t)Np * Kp;
  pack_weights_kernel<<<(unsigned)((wn + 255) / 256), 256, 0, st>>>(L.w, L.cout, L.cin, W.Wp, Np, Kp);
  dim3 grid((unsigned)((Mp + 127) / 128), (unsigned)(Np / 64));
  wmma_gemm_bias_kernel<<<grid, 256, 0, st>>>(Xin, W.Wp, L.b, W.Y, Mp, Kp, Np, L.cout);
  zero_stats_kernel<<<4, 256, 0, st>>>(W.sums, W.sqs, 1024);
  bn_partial_kernel<<<256, 256, 0, st>>>(W.Y, M, Np, L.cout, W.sums, W.sqs);
  bn_finalize_kernel<<<(L.cout + 255) / 256, 256, 0, st>>>(W.sums, W.sqs, M, L.cout, L.g, L.bt, W.sc, W.sh);
  size_t tot = (size_t)Mp * KpO;
  bn_relu_store_f16_kernel<<<(unsigned)((tot + 255) / 256), 256, 0, st>>>(W.Y, M, Np, L.cout, W.sc, W.sh, Xout, Mp, KpO);
}

void run_sa_stage(int cfg, const LayerP (*Ls)[3], int nbr,
                  const float* xyz_in, int Nin, const float* feat_in, int featC,
                  float* xyz_out, float* feat_out, int outC,
                  int* idx, int* gidx, _Float16* Xa, _Float16* Xb,
                  const Scratch& W, hipStream_t st)
{
  const int B = 2;
  int S = SA_NPOINT[cfg];
  size_t smem = (size_t)Nin * 4 + 256 * 8;
  fps_kernel<<<B, 256, smem, st>>>(xyz_in, Nin, S, idx);
  int tot = B * S;
  gather_xyz_kernel<<<(tot + 255) / 256, 256, 0, st>>>(xyz_in, idx, Nin, S, xyz_out);
  int coff = 0;
  for (int br = 0; br < nbr; ++br) {
    float r = SA_RADII[cfg][br];
    int K = SA_NS[cfg][br];
    ball_query_kernel<<<(tot + 127) / 128, 128, 0, st>>>(xyz_in, xyz_out, Nin, S, r * r, K, gidx);
    int M = B * S * K, Mp = ru16(M);
    int Cin = featC + 3, Kp = ru32(Cin);
    group_pack_kernel<<<(Mp + 127) / 128, 128, 0, st>>>(xyz_in, xyz_out, feat_in, featC,
                                                        gidx, Nin, S, K, Cin, Kp, Xa, Mp, M);
    const _Float16* cur = Xa;
    _Float16* nxt = Xb;
    for (int l = 0; l < 3; ++l) {
      run_layer(Ls[br][l], cur, M, Mp, W, nxt, st);
      const _Float16* t = cur; cur = nxt; nxt = const_cast<_Float16*>(t);
    }
    int C2 = Ls[br][2].cout, KpA = ru32(C2);
    int tp = B * S * C2;
    maxpool_kernel<<<(tp + 255) / 256, 256, 0, st>>>(cur, B * S, K, KpA, C2, feat_out, outC, coff);
    coff += C2;
  }
}

void run_fp_stack(const float* xyz1, int N1, const float* xyz2, int N2,
                  const float* pts1, int C1, const float* pts2, int C2,
                  const LayerP* Ls, float* feat_out,
                  _Float16* Xa, _Float16* Xb, const Scratch& W, hipStream_t st,
                  const _Float16** last_f16, int* last_Mp)
{
  const int B = 2;
  int M = B * N1, Mp = ru16(M);
  int Cin = C1 + C2, Kp = ru32(Cin);
  fp_interp_pack_kernel<<<(Mp + 127) / 128, 128, 0, st>>>(xyz1, xyz2, pts1, C1, pts2, C2,
                                                          N1, N2, Xa, Kp, Mp, M);
  const _Float16* cur = Xa;
  _Float16* nxt = Xb;
  for (int l = 0; l < 2; ++l) {
    run_layer(Ls[l], cur, M, Mp, W, nxt, st);
    const _Float16* t = cur; cur = nxt; nxt = const_cast<_Float16*>(t);
  }
  const LayerP& LL = Ls[1];
  int Np = ru64i(LL.cout);
  size_t tot = (size_t)M * LL.cout;
  bn_relu_store_f32_kernel<<<(unsigned)((tot + 255) / 256), 256, 0, st>>>(W.Y, M, Np, LL.cout, W.sc, W.sh, feat_out);
  if (last_f16) { *last_f16 = cur; *last_Mp = Mp; }
}

} // namespace

extern "C" void kernel_launch(void* const* d_in, const int* in_sizes, int n_in,
                              void* d_out, int out_size, void* d_ws, size_t ws_size,
                              hipStream_t stream)
{
  (void)in_sizes; (void)n_in; (void)out_size; (void)ws_size;
  const int B = 2, N0 = 2048;
  const float* xyz_in = (const float*)d_in[0];
  const float* cls    = (const float*)d_in[1];

  // --- walk params in setup_inputs() dict order ---
  int pi = 2;
  LayerP sa[4][5][3];
  for (int l = 0; l < 4; ++l) {
    for (int br = 0; br < SA_NBR[l]; ++br) {
      int cin = SA_CIN[l];
      for (int ly = 0; ly < 3; ++ly) {
        LayerP& L = sa[l][br][ly];
        L.w = (const float*)d_in[pi++]; L.b  = (const float*)d_in[pi++];
        L.g = (const float*)d_in[pi++]; L.bt = (const float*)d_in[pi++];
        L.cin = cin; L.cout = SA_DIMS[l][br][ly]; cin = L.cout;
      }
    }
  }
  LayerP fpl[5][2]; // fp5, fp4, fp3, fp2, fp1
  for (int k = 0; k < 5; ++k) {
    int cin = FP_CIN[k];
    for (int ly = 0; ly < 2; ++ly) {
      LayerP& L = fpl[k][ly];
      L.w = (const float*)d_in[pi++]; L.b  = (const float*)d_in[pi++];
      L.g = (const float*)d_in[pi++]; L.bt = (const float*)d_in[pi++];
      L.cin = cin; L.cout = FP_DIMS[k][ly]; cin = L.cout;
    }
  }
  LayerP head1;
  head1.w = (const float*)d_in[pi++]; head1.b  = (const float*)d_in[pi++];
  head1.g = (const float*)d_in[pi++]; head1.bt = (const float*)d_in[pi++];
  head1.cin = 128; head1.cout = 128;
  const float* W2 = (const float*)d_in[pi++];
  const float* b2 = (const float*)d_in[pi++];

  // --- workspace bump allocation ---
  char* base = (char*)d_ws;
  size_t off = 0;
  auto A = [&](size_t bytes) -> void* {
    void* p = base + off;
    off = (off + bytes + 255) & ~(size_t)255;
    return p;
  };
  float* l0x = (float*)A((size_t)B * 2048 * 3 * 4);
  float* l1x = (float*)A((size_t)B * 1024 * 3 * 4);
  float* l2x = (float*)A((size_t)B * 512 * 3 * 4);
  float* l3x = (float*)A((size_t)B * 256 * 3 * 4);
  float* l4x = (float*)A((size_t)B * 128 * 3 * 4);
  float* l5x = (float*)A((size_t)B * 128 * 3 * 4);
  float* l1f = (float*)A((size_t)B * 1024 * 512 * 4);
  float* l2f = (float*)A((size_t)B * 512 * 1280 * 4);
  float* l3f = (float*)A((size_t)B * 256 * 1024 * 4);
  float* l4f = (float*)A((size_t)B * 128 * 1024 * 4);
  float* l5f = (float*)A((size_t)B * 128 * 1024 * 4);
  float* l4n = (float*)A((size_t)B * 128 * 256 * 4);
  float* l3n = (float*)A((size_t)B * 256 * 256 * 4);
  float* l2n = (float*)A((size_t)B * 512 * 128 * 4);
  float* l1n = (float*)A((size_t)B * 1024 * 128 * 4);
  float* l0n = (float*)A((size_t)B * 2048 * 128 * 4);
  float* p0  = (float*)A((size_t)B * 2048 * 7 * 4);
  int*   idx  = (int*)A((size_t)B * 1024 * 4);
  int*   gidx = (int*)A((size_t)B * 1024 * 384 * 4);
  Scratch W;
  W.sc   = (float*)A(1024 * 4);
  W.sh   = (float*)A(1024 * 4);
  W.sums = (float*)A(1024 * 4);
  W.sqs  = (float*)A(1024 * 4);
  W.Wp   = (_Float16*)A((size_t)2 * 1024 * 1024 * 2);
  const size_t XE = (size_t)200704 * 544; // max padded activation elements
  _Float16* Xa = (_Float16*)A(XE * 2);
  _Float16* Xb = (_Float16*)A(XE * 2);
  W.Y = (float*)A((size_t)786432 * 128 * 4);

  // --- forward graph ---
  transpose_xyz_kernel<<<(B * N0 * 3 + 255) / 256, 256, 0, stream>>>(xyz_in, N0, l0x);

  run_sa_stage(0, sa[0], 5, l0x, 2048, l0x, 3,    l1x, l1f, 512,  idx, gidx, Xa, Xb, W, stream);
  run_sa_stage(1, sa[1], 5, l1x, 1024, l1f, 512,  l2x, l2f, 1280, idx, gidx, Xa, Xb, W, stream);
  run_sa_stage(2, sa[2], 4, l2x, 512,  l2f, 1280, l3x, l3f, 1024, idx, gidx, Xa, Xb, W, stream);
  run_sa_stage(3, sa[3], 4, l3x, 256,  l3f, 1024, l4x, l4f, 1024, idx, gidx, Xa, Xb, W, stream);
  run_sa_stage(3, sa[3], 4, l4x, 128,  l4f, 1024, l5x, l5f, 1024, idx, gidx, Xa, Xb, W, stream);

  run_fp_stack(l4x, 128,  l5x, 128, l4f, 1024, l5f, 1024, fpl[0], l4n, Xa, Xb, W, stream, nullptr, nullptr);
  run_fp_stack(l3x, 256,  l4x, 128, l3f, 1024, l4n, 256,  fpl[1], l3n, Xa, Xb, W, stream, nullptr, nullptr);
  run_fp_stack(l2x, 512,  l3x, 256, l2f, 1280, l3n, 256,  fpl[2], l2n, Xa, Xb, W, stream, nullptr, nullptr);
  run_fp_stack(l1x, 1024, l2x, 512, l1f, 512,  l2n, 128,  fpl[3], l1n, Xa, Xb, W, stream, nullptr, nullptr);

  build_p0_kernel<<<(B * N0 + 255) / 256, 256, 0, stream>>>(cls, l0x, N0, p0);
  const _Float16* l0f16 = nullptr;
  int Mp0 = 0;
  run_fp_stack(l0x, 2048, l1x, 1024, p0, 7, l1n, 128, fpl[4], l0n, Xa, Xb, W, stream, &l0f16, &Mp0);

  // head: conv1 (128->128, BN+ReLU via WMMA path) into Xb, then logits + log_softmax
  run_layer(head1, l0f16, B * N0, Mp0, W, Xb, stream);
  float* out = (float*)d_out;
  head_logits_kernel<<<(B * N0 + 255) / 256, 256, 0, stream>>>(Xb, 128, W2, b2, B * N0, out);

  // second output: transpose(post-fp4 l3_points) -> [B, 256, 256]
  transpose_feat_kernel<<<(B * 256 * 256 + 255) / 256, 256, 0, stream>>>(l3n, 256, 256, out + (size_t)B * N0 * 2);
}
